// Attention_77309411618
// MI455X (gfx1250) — compile-verified, matrix operand order checked by reading, not run on hardware
//
#include <hip/hip_runtime.h>
#include <hip/hip_bf16.h>

typedef __attribute__((ext_vector_type(16))) _Float16 v16h;
typedef __attribute__((ext_vector_type(8)))  float    v8f;

#define B_  2
#define N_  400
#define L_  16384
#define C_  256
#define H_  8
#define D_  32
#define QT_ 25            // N_/16
#define LSPLIT 16
#define SEG (L_/LSPLIT)   // 1024 keys per split
#define LDSPITCH 264      // padded row pitch (halfs): 132 dwords -> bank stride 4

// ---------------------------------------------------------------------------
// WMMA helper: D = A(16x32 f16) * B(32x16 f16) + C(16x16 f32)
static __device__ __forceinline__ v8f wmma16(v16h a, v16h b, v8f c) {
  return __builtin_amdgcn_wmma_f32_16x16x32_f16(
      /*neg_a=*/false, a, /*neg_b=*/false, b,
      /*c_mod=*/(short)0, c, /*reuse_a=*/false, /*reuse_b=*/false);
}

static __device__ __forceinline__ v8f vzero8() {
  v8f z;
#pragma unroll
  for (int i = 0; i < 8; ++i) z[i] = 0.0f;
  return z;
}

// Load one 16x32 tile in CDNA5 WMMA 16-bit A/B layout from a row-major f16 matrix.
// lane<16 : elements K{0..7} and K{16..23} ; lane>=16: K{8..15} and K{24..31}
static __device__ __forceinline__ v16h ld_tile_f16(const _Float16* X, size_t ld,
                                                   size_t row, size_t k0, int lane) {
  const _Float16* p = X + row * ld + k0 + (size_t)((lane >> 4) << 3);
  v16h a;
#pragma unroll
  for (int i = 0; i < 8; ++i) {
    a[i]     = p[i];
    a[8 + i] = p[16 + i];
  }
  return a;
}

// Same tile pattern out of an LDS-resident padded panel.
static __device__ __forceinline__ v16h ld_tile_lds(const _Float16* As,
                                                   int row, int k0, int lane) {
  const _Float16* p = As + row * LDSPITCH + k0 + ((lane >> 4) << 3);
  v16h a;
#pragma unroll
  for (int i = 0; i < 8; ++i) {
    a[i]     = p[i];
    a[8 + i] = p[16 + i];
  }
  return a;
}

// ---------------------------------------------------------------------------
// One-shot weight convert: 256x256 f32 -> f16 (row-major preserved).
__global__ void wcvt_kernel(const float* __restrict__ W, _Float16* __restrict__ out) {
  const int i = blockIdx.x * blockDim.x + threadIdx.x;   // exactly 65536 threads
  out[i] = (_Float16)W[i];
}

// ---------------------------------------------------------------------------
// Generic 256-wide projection: out = X[rows x 256] @ Wh^T  (+bias), Wh is f16.
// A panel (16 rows x 256) staged once per block in LDS as f16.
// mode 0: f16 row-major, value*scale          (q/k projections)
// mode 1: f16 transposed [256][out_ld]        (v projection -> vT)
// mode 2: f32 row-major + bias                (final output projection)
// block = 128 (4 waves); each wave: 16 rows x 64 cols; grid.x = rows/16
__global__ void proj_gemm_kernel(const float* __restrict__ X,
                                 const _Float16* __restrict__ Wh,
                                 const float* __restrict__ bias,
                                 float scale, int mode,
                                 void* __restrict__ out, size_t out_ld) {
  __shared__ _Float16 As[16 * LDSPITCH];    // 8448 B

  const int lane = threadIdx.x & 31;
  const int wave = threadIdx.x >> 5;
  const size_t row0 = (size_t)blockIdx.x * 16;
  const int n0 = wave * 64;

  // cooperative A-panel fill: 8 threads per row, 32 consecutive cols each
  {
    const int r  = threadIdx.x >> 3;
    const int c0 = (threadIdx.x & 7) * 32;
    const float* src = X + (row0 + r) * C_ + c0;
    _Float16* dst = As + r * LDSPITCH + c0;
#pragma unroll
    for (int i = 0; i < 32; ++i) dst[i] = (_Float16)src[i];
  }
  __syncthreads();

  v8f acc[4];
#pragma unroll
  for (int j = 0; j < 4; ++j) acc[j] = vzero8();

#pragma unroll
  for (int k0 = 0; k0 < C_; k0 += 32) {
    v16h a = ld_tile_lds(As, lane & 15, k0, lane);
#pragma unroll
    for (int j = 0; j < 4; ++j) {
      v16h bt = ld_tile_f16(Wh, C_, (size_t)(n0 + 16 * j + (lane & 15)), k0, lane);
      acc[j] = wmma16(a, bt, acc[j]);
    }
  }

  const int nn   = lane & 15;
  const int mofs = (lane >> 4) << 3;
#pragma unroll
  for (int j = 0; j < 4; ++j) {
    const int c = n0 + 16 * j + nn;
#pragma unroll
    for (int r = 0; r < 8; ++r) {
      const size_t m = row0 + (size_t)(r + mofs);
      const float v = acc[j][r];
      if (mode == 0)       ((_Float16*)out)[m * C_ + c]             = (_Float16)(v * scale);
      else if (mode == 1)  ((_Float16*)out)[(size_t)c * out_ld + m] = (_Float16)v;
      else                 ((float*)out)[m * C_ + c]                = v + bias[c];
    }
  }
}

// ---------------------------------------------------------------------------
// Mask kernel: per (b, q-tile, 16-key tile) compute pre-softmax scores for all
// 8 heads (standard orientation: D lane = key -> coalesced mask writes), run
// the head-mixing MLP per element, write sigmoid mask [B][N][L].
// block = 128 (4 waves, one 16-key tile each); grid = (L/64, QT, B)
__global__ void mask_kernel(const _Float16* __restrict__ qh,
                            const _Float16* __restrict__ kh,
                            const unsigned char* __restrict__ kpm,
                            const float* __restrict__ W1, const float* __restrict__ b1,
                            const float* __restrict__ W2, const float* __restrict__ b2,
                            float* __restrict__ mask_out) {
  const int lane = threadIdx.x & 31;
  const int wave = threadIdx.x >> 5;
  const int b  = blockIdx.z;
  const int n0 = blockIdx.y * 16;
  const int l0 = (blockIdx.x * 4 + wave) * 16;

  // A = Q tiles (16q x 32d) for all 8 heads; rows of qh are contiguous.
  v16h qa[H_];
#pragma unroll
  for (int h = 0; h < H_; ++h)
    qa[h] = ld_tile_f16(qh, C_, (size_t)b * N_ + n0 + (lane & 15), (size_t)h * D_, lane);

  // B = K^T tiles: lane n = key holds row (l0+n) of kh -> contiguous.
  v8f s[H_];
#pragma unroll
  for (int h = 0; h < H_; ++h) {
    v16h kb = ld_tile_f16(kh, C_, (size_t)b * L_ + l0 + (lane & 15), (size_t)h * D_, lane);
    s[h] = wmma16(qa[h], kb, vzero8());
  }

  const bool pad = kpm[(size_t)b * L_ + l0 + (lane & 15)] != 0;
  const int l    = l0 + (lane & 15);
  const int mofs = (lane >> 4) << 3;

#pragma unroll
  for (int r = 0; r < 8; ++r) {
    float sh[H_];
#pragma unroll
    for (int h = 0; h < H_; ++h) sh[h] = pad ? -__builtin_inff() : s[h][r];
    float o = b2[0];
#pragma unroll
    for (int j = 0; j < H_; ++j) {
      float hj = b1[j];
#pragma unroll
      for (int h = 0; h < H_; ++h) hj = fmaf(W1[j * H_ + h], sh[h], hj);
      hj = fmaxf(hj, 0.0f);
      o = fmaf(W2[j], hj, o);
    }
    const float mval = 1.0f / (1.0f + __expf(-o));
    const int q = n0 + r + mofs;
    mask_out[((size_t)b * N_ + q) * L_ + l] = mval;  // lanes 0..15 -> 64B coalesced
  }
}

// ---------------------------------------------------------------------------
// Flash attention, split-L. 1 wave = 1 head; block = 256 (8 heads);
// grid = (B*QT, LSPLIT). Scores computed transposed (S^T = K*Q^T) so the
// exp(P) tile is already in WMMA A-layout for P@V with zero lane traffic.
__global__ void flash_kernel(const _Float16* __restrict__ qh,
                             const _Float16* __restrict__ kh,
                             const _Float16* __restrict__ vT,
                             const unsigned char* __restrict__ kpm,
                             float* __restrict__ Opart,
                             float* __restrict__ Mst, float* __restrict__ Lst) {
  const int lane = threadIdx.x & 31;
  const int h    = threadIdx.x >> 5;
  const int b    = blockIdx.x / QT_;
  const int qt   = blockIdx.x % QT_;
  const int split = blockIdx.y;
  const int lbeg  = split * SEG;
  const int hi8   = (lane >> 4) << 3;

  // B = Q^T (32d x 16q): lane q holds row q of qh (contiguous)
  const v16h qb = ld_tile_f16(qh, C_, (size_t)b * N_ + qt * 16 + (lane & 15),
                              (size_t)h * D_, lane);

  v8f o0 = vzero8(), o1 = vzero8();
  float m_run = -1.0e30f, l_run = 0.0f;
  const int srcbase = (lane & 16) >> 1;  // 0 or 8: q-index base for this half

  for (int lk = lbeg; lk < lbeg + SEG; lk += 32) {
    // prefetch next chunk of K and V streams (sequential out of L2)
    if (lk + 32 < lbeg + SEG) {
      __builtin_prefetch((const void*)(kh + ((size_t)b * L_ + lk + 32 + (lane & 15)) * C_ +
                                       (size_t)h * D_), 0, 1);
      __builtin_prefetch((const void*)(vT + ((size_t)h * D_ + (lane & 15)) * ((size_t)B_ * L_) +
                                       (size_t)b * L_ + lk + 32), 0, 1);
    }

    // A = K rows (16key x 32d), two key sub-tiles
    v16h ka0 = ld_tile_f16(kh, C_, (size_t)b * L_ + lk + (lane & 15), (size_t)h * D_, lane);
    v16h ka1 = ld_tile_f16(kh, C_, (size_t)b * L_ + lk + 16 + (lane & 15), (size_t)h * D_, lane);
    v8f s0 = wmma16(ka0, qb, vzero8());  // S^T: lane=q, VGPR r=key
    v8f s1 = wmma16(ka1, qb, vzero8());

    // key padding mask (uniform per VGPR row within half-wave)
    const unsigned char* kp = kpm + (size_t)b * L_ + lk + hi8;
    unsigned long long g0 = *(const unsigned long long*)(kp);
    unsigned long long g1 = *(const unsigned long long*)(kp + 16);
#pragma unroll
    for (int r = 0; r < 8; ++r) {
      if ((g0 >> (8 * r)) & 1ull) s0[r] = -__builtin_inff();
      if ((g1 >> (8 * r)) & 1ull) s1[r] = -__builtin_inff();
    }

    // online softmax: per-q row stats live in lanes q and q+16
    float cm = -__builtin_inff();
#pragma unroll
    for (int r = 0; r < 8; ++r) cm = fmaxf(cm, fmaxf(s0[r], s1[r]));
    cm = fmaxf(cm, __shfl_xor(cm, 16));
    const float m_new = fmaxf(m_run, cm);
    const float alpha = __expf(m_run - m_new);

    float rs = 0.0f;
    v16h pa;  // exp(P) lands directly in WMMA A-layout (the S^T trick)
#pragma unroll
    for (int r = 0; r < 8; ++r) {
      const float e0 = __expf(s0[r] - m_new);
      const float e1 = __expf(s1[r] - m_new);
      rs += e0 + e1;
      pa[r]     = (_Float16)e0;
      pa[8 + r] = (_Float16)e1;
    }
    rs += __shfl_xor(rs, 16);
    l_run = l_run * alpha + rs;
    m_run = m_new;

    // rescale O accumulators: broadcast per-q alpha (O layout: VGPR r -> q)
#pragma unroll
    for (int r = 0; r < 8; ++r) {
      const float f = __shfl(alpha, srcbase + r);
      o0[r] *= f;
      o1[r] *= f;
    }

    // B = V (32key x 16dd) from vT[C][B*L]: contiguous per lane along keys
    v16h vb0 = ld_tile_f16(vT, (size_t)B_ * L_, (size_t)h * D_ + (lane & 15),
                           (size_t)b * L_ + lk, lane);
    v16h vb1 = ld_tile_f16(vT, (size_t)B_ * L_, (size_t)h * D_ + 16 + (lane & 15),
                           (size_t)b * L_ + lk, lane);
    o0 = wmma16(pa, vb0, o0);
    o1 = wmma16(pa, vb1, o1);
  }

  // write split partials + stats
  const size_t sb = ((((size_t)b * QT_ + qt) * H_ + h) * LSPLIT + split);
  float* op = Opart + sb * (16 * 32);
  const int nn   = lane & 15;
  const int mofs = (lane >> 4) << 3;
#pragma unroll
  for (int r = 0; r < 8; ++r) {
    const int q = r + mofs;
    op[q * 32 + nn]      = o0[r];
    op[q * 32 + 16 + nn] = o1[r];
  }
  if (lane < 16) {
    Mst[sb * 16 + lane] = m_run;
    Lst[sb * 16 + lane] = l_run;
  }
}

// ---------------------------------------------------------------------------
// Merge LSPLIT partials with log-sum-exp, write pre-projection activations.
// grid = (B*QT, H); block = 512 (thread = q*32 + dd)
__global__ void combine_kernel(const float* __restrict__ Opart,
                               const float* __restrict__ Mst,
                               const float* __restrict__ Lst,
                               float* __restrict__ xacc) {
  const int b  = blockIdx.x / QT_;
  const int qt = blockIdx.x % QT_;
  const int h  = blockIdx.y;
  const int q  = threadIdx.x >> 5;
  const int dd = threadIdx.x & 31;
  const size_t sb = (((size_t)b * QT_ + qt) * H_ + h) * LSPLIT;

  float M = -__builtin_inff();
#pragma unroll
  for (int s = 0; s < LSPLIT; ++s) M = fmaxf(M, Mst[(sb + s) * 16 + q]);
  float Ls = 0.0f, Ov = 0.0f;
#pragma unroll
  for (int s = 0; s < LSPLIT; ++s) {
    const float w = __expf(Mst[(sb + s) * 16 + q] - M);
    Ls += Lst[(sb + s) * 16 + q] * w;
    Ov += Opart[(sb + s) * (16 * 32) + q * 32 + dd] * w;
  }
  xacc[((size_t)b * N_ + qt * 16 + q) * C_ + h * D_ + dd] = Ov / Ls;
}

// ---------------------------------------------------------------------------
extern "C" void kernel_launch(void* const* d_in, const int* in_sizes, int n_in,
                              void* d_out, int out_size, void* d_ws, size_t ws_size,
                              hipStream_t stream) {
  const float* query = (const float*)d_in[0];
  const float* key   = (const float*)d_in[1];
  const float* value = (const float*)d_in[2];
  const unsigned char* kpm = (const unsigned char*)d_in[3];  // bool mask
  const float* Wq = (const float*)d_in[4];
  const float* Wk = (const float*)d_in[5];
  const float* Wv = (const float*)d_in[6];
  const float* Wp = (const float*)d_in[7];
  const float* bp = (const float*)d_in[8];
  const float* W1 = (const float*)d_in[9];
  const float* b1 = (const float*)d_in[10];
  const float* W2 = (const float*)d_in[11];
  const float* b2 = (const float*)d_in[12];

  float* xout     = (float*)d_out;                       // [B,N,C]
  float* mask_out = xout + (size_t)B_ * N_ * C_;         // [B,N,L]

  char* ws = (char*)d_ws;
  size_t off = 0;
  _Float16* qh = (_Float16*)(ws + off); off += (size_t)B_ * N_ * C_ * 2;
  _Float16* kh = (_Float16*)(ws + off); off += (size_t)B_ * L_ * C_ * 2;
  _Float16* vT = (_Float16*)(ws + off); off += (size_t)B_ * L_ * C_ * 2;  // [C][B*L]
  float* xacc  = (float*)(ws + off);    off += (size_t)B_ * N_ * C_ * 4;
  float* Opart = (float*)(ws + off);    off += (size_t)B_ * QT_ * H_ * LSPLIT * 16 * 32 * 4;
  float* Mst   = (float*)(ws + off);    off += (size_t)B_ * QT_ * H_ * LSPLIT * 16 * 4;
  float* Lst   = (float*)(ws + off);    off += (size_t)B_ * QT_ * H_ * LSPLIT * 16 * 4;
  _Float16* Wqh = (_Float16*)(ws + off); off += (size_t)C_ * C_ * 2;
  _Float16* Wkh = (_Float16*)(ws + off); off += (size_t)C_ * C_ * 2;
  _Float16* Wvh = (_Float16*)(ws + off); off += (size_t)C_ * C_ * 2;
  _Float16* Wph = (_Float16*)(ws + off); off += (size_t)C_ * C_ * 2;

  const float scale = 0.17677669529663687f;  // 1/sqrt(d), folded into q

  // one-shot weight conversions (1 MB f32 in, 512 KB f16 out -> negligible)
  wcvt_kernel<<<dim3((C_ * C_) / 256), 256, 0, stream>>>(Wq, Wqh);
  wcvt_kernel<<<dim3((C_ * C_) / 256), 256, 0, stream>>>(Wk, Wkh);
  wcvt_kernel<<<dim3((C_ * C_) / 256), 256, 0, stream>>>(Wv, Wvh);
  wcvt_kernel<<<dim3((C_ * C_) / 256), 256, 0, stream>>>(Wp, Wph);

  proj_gemm_kernel<<<dim3((B_ * N_) / 16), 128, 0, stream>>>(query, Wqh, bp, scale, 0, qh, 0);
  proj_gemm_kernel<<<dim3((B_ * L_) / 16), 128, 0, stream>>>(key,   Wkh, bp, 1.0f,  0, kh, 0);
  proj_gemm_kernel<<<dim3((B_ * L_) / 16), 128, 0, stream>>>(value, Wvh, bp, 1.0f,  1, vT, (size_t)B_ * L_);
  mask_kernel<<<dim3(L_ / 64, QT_, B_), 128, 0, stream>>>(qh, kh, kpm, W1, b1, W2, b2, mask_out);
  flash_kernel<<<dim3(B_ * QT_, LSPLIT), 256, 0, stream>>>(qh, kh, vT, kpm, Opart, Mst, Lst);
  combine_kernel<<<dim3(B_ * QT_, H_), 512, 0, stream>>>(Opart, Mst, Lst, xacc);
  proj_gemm_kernel<<<dim3((B_ * N_) / 16), 128, 0, stream>>>(xacc, Wph, bp, 1.0f, 2, xout, 0);
}
